// Attention_42709154791642
// MI455X (gfx1250) — matrix operand order repaired and checked by reading, not compiled, not run on hardware
//
#include <hip/hip_runtime.h>

// ---------------- types ----------------
typedef __attribute__((ext_vector_type(16))) __bf16       v16bf;
typedef __attribute__((ext_vector_type(2)))  __bf16       v2bf;
typedef __attribute__((ext_vector_type(8)))  float        v8f;
typedef __attribute__((ext_vector_type(4)))  unsigned int uint32x4;

struct __align__(16) FragBits { uint32x4 lo, hi; };

// native bf16 convert (v_cvt_pk_bf16_f32 on gfx1250; RNE)
__device__ __forceinline__ unsigned short f2bf(float x) {
  __bf16 h = (__bf16)x;
  return __builtin_bit_cast(unsigned short, h);
}
__device__ __forceinline__ unsigned int pack2(float a, float b) {
  v2bf p; p.x = (__bf16)a; p.y = (__bf16)b;
  return __builtin_bit_cast(unsigned int, p);
}

// problem constants
#define NNq  1024
#define EEc  512
#define HHc  8
#define T3c  3072
#define NEc  (1024LL * 512)
#define T3Ec (3072LL * 512)
#define NT3c (1024LL * 3072)
#define NBX  24            // 3072/128 kv tiles
#define NBY  16            // 1024/64 q tiles
#define NPART (NBX * NBY)  // 384 score partials per (b,h)

// ---------------------------------------------------------------------------
// Generic batched strided NT GEMM (exact tiles), f32 in/out, bf16 WMMA.
//   C[m,n] = sum_k A[m*rsA + k*csA] * B[n*rsB + k*csB]
// 256 threads = 8 waves; WG tile 64x128; K-chunk 64; software pipelined.
// ---------------------------------------------------------------------------
template <bool VECA, bool VECB>
__global__ __launch_bounds__(256) void gemm_bf16_nt(
    const float* __restrict__ A, long long rsA, long long csA, long long sA1, long long sA2,
    const float* __restrict__ Bm, long long rsB, long long csB, long long sB1, long long sB2,
    float* __restrict__ C, long long ldC, long long sC1, long long sC2,
    int M, int Nn, int K, int n2)
{
  __shared__ __align__(16) unsigned short As[64][72];
  __shared__ __align__(16) unsigned short Bs[128][72];

  const int tid = threadIdx.x;
  const int z  = blockIdx.z;
  const int b1 = z / n2, b2 = z - b1 * n2;
  const int tileM = blockIdx.y * 64;
  const int tileN = blockIdx.x * 128;

  const float* Ab = A  + (long long)b1 * sA1 + (long long)b2 * sA2 + (long long)tileM * rsA;
  const float* Bb = Bm + (long long)b1 * sB1 + (long long)b2 * sB2 + (long long)tileN * rsB;
  float*       Cb = C  + (long long)b1 * sC1 + (long long)b2 * sC2;

  const int wid  = tid >> 5, lane = tid & 31, lr = lane & 15, half = lane >> 4;
  const int wm   = (wid >> 2) * 32, wn = (wid & 3) * 32;

  const v8f vzero = {0.f,0.f,0.f,0.f,0.f,0.f,0.f,0.f};
  v8f acc[2][2];
  acc[0][0] = vzero; acc[0][1] = vzero; acc[1][0] = vzero; acc[1][1] = vzero;

  float4 aV[4], bV[8];
  float  aS[16], bS[32];

  auto loadA = [&](int k0) {
    if (VECA) {
      #pragma unroll
      for (int i = 0; i < 4; ++i) {
        int idx = tid + i * 256; int r = idx >> 4, c4 = idx & 15;
        aV[i] = *(const float4*)(Ab + (long long)r * rsA + (k0 + c4 * 4));
      }
    } else {
      #pragma unroll
      for (int i = 0; i < 16; ++i) {
        int idx = tid + i * 256; int r = idx & 63, c = idx >> 6;
        aS[i] = Ab[(long long)r * rsA + (long long)(k0 + c) * csA];
      }
    }
  };
  auto loadB = [&](int k0) {
    if (VECB) {
      #pragma unroll
      for (int i = 0; i < 8; ++i) {
        int idx = tid + i * 256; int r = idx >> 4, c4 = idx & 15;
        bV[i] = *(const float4*)(Bb + (long long)r * rsB + (k0 + c4 * 4));
      }
    } else {
      #pragma unroll
      for (int i = 0; i < 32; ++i) {
        int idx = tid + i * 256; int r = idx & 127, c = idx >> 7;
        bS[i] = Bb[(long long)r * rsB + (long long)(k0 + c) * csB];
      }
    }
  };
  auto storeA = [&]() {
    if (VECA) {
      #pragma unroll
      for (int i = 0; i < 4; ++i) {
        int idx = tid + i * 256; int r = idx >> 4, c4 = idx & 15;
        uint2 u; u.x = pack2(aV[i].x, aV[i].y); u.y = pack2(aV[i].z, aV[i].w);
        *(uint2*)&As[r][c4 * 4] = u;
      }
    } else {
      #pragma unroll
      for (int i = 0; i < 16; ++i) {
        int idx = tid + i * 256; int r = idx & 63, c = idx >> 6;
        As[r][c] = f2bf(aS[i]);
      }
    }
  };
  auto storeB = [&]() {
    if (VECB) {
      #pragma unroll
      for (int i = 0; i < 8; ++i) {
        int idx = tid + i * 256; int r = idx >> 4, c4 = idx & 15;
        uint2 u; u.x = pack2(bV[i].x, bV[i].y); u.y = pack2(bV[i].z, bV[i].w);
        *(uint2*)&Bs[r][c4 * 4] = u;
      }
    } else {
      #pragma unroll
      for (int i = 0; i < 32; ++i) {
        int idx = tid + i * 256; int r = idx & 127, c = idx >> 7;
        Bs[r][c] = f2bf(bS[i]);
      }
    }
  };

  loadA(0); loadB(0);
  for (int k0 = 0; k0 < K; k0 += 64) {
    storeA(); storeB();
    __syncthreads();
    if (k0 + 64 < K) { loadA(k0 + 64); loadB(k0 + 64); }

    #pragma unroll
    for (int s2 = 0; s2 < 2; ++s2) {
      const int kc = s2 * 32;
      v16bf a[2], b[2];
      #pragma unroll
      for (int i = 0; i < 2; ++i) {
        int row = wm + i * 16 + lr;
        FragBits f;
        f.lo = *(const uint32x4*)&As[row][kc + half * 8];
        f.hi = *(const uint32x4*)&As[row][kc + 16 + half * 8];
        a[i] = __builtin_bit_cast(v16bf, f);
      }
      #pragma unroll
      for (int j = 0; j < 2; ++j) {
        int row = wn + j * 16 + lr;
        FragBits f;
        f.lo = *(const uint32x4*)&Bs[row][kc + half * 8];
        f.hi = *(const uint32x4*)&Bs[row][kc + 16 + half * 8];
        b[j] = __builtin_bit_cast(v16bf, f);
      }
      #pragma unroll
      for (int i = 0; i < 2; ++i)
        #pragma unroll
        for (int j = 0; j < 2; ++j)
          acc[i][j] = __builtin_amdgcn_wmma_f32_16x16x32_bf16(
              false, a[i], false, b[j], (short)0, acc[i][j], false, false);
    }
    __syncthreads();
  }

  #pragma unroll
  for (int i = 0; i < 2; ++i)
    #pragma unroll
    for (int j = 0; j < 2; ++j)
      #pragma unroll
      for (int r = 0; r < 8; ++r) {
        int row = tileM + wm + i * 16 + half * 8 + r;
        int col = tileN + wn + j * 16 + lr;
        Cb[(long long)row * ldC + col] = acc[i][j][r];
      }
}

// ---------------------------------------------------------------------------
// Channel-attention stats (two-pass, deterministic) + fused norm-softmax.
// ---------------------------------------------------------------------------
#define NCHUNK 64

__global__ __launch_bounds__(256) void stat_partial(
    const float* __restrict__ x, long long count, double* __restrict__ partials)
{
  __shared__ double s0[256], s1[256];
  const int chunk = blockIdx.x, batch = blockIdx.y, tid = threadIdx.x;
  const long long base = (long long)batch * count;
  const long long clen = (count + NCHUNK - 1) / NCHUNK;
  const long long lo = (long long)chunk * clen;
  const long long hi = (lo + clen < count) ? lo + clen : count;
  double s = 0.0, ss = 0.0;
  for (long long i = lo + tid; i < hi; i += 256) {
    float v = x[base + i];
    s += (double)v; ss += (double)v * (double)v;
  }
  s0[tid] = s; s1[tid] = ss; __syncthreads();
  for (int st = 128; st > 0; st >>= 1) {
    if (tid < st) { s0[tid] += s0[tid + st]; s1[tid] += s1[tid + st]; }
    __syncthreads();
  }
  if (tid == 0) {
    partials[((long long)batch * NCHUNK + chunk) * 2 + 0] = s0[0];
    partials[((long long)batch * NCHUNK + chunk) * 2 + 1] = s1[0];
  }
}

__global__ __launch_bounds__(64) void stat_final(
    const double* __restrict__ partials, long long count, float* __restrict__ stats)
{
  __shared__ double s0[64], s1[64];
  const int batch = blockIdx.x, tid = threadIdx.x;
  s0[tid] = partials[((long long)batch * NCHUNK + tid) * 2 + 0];
  s1[tid] = partials[((long long)batch * NCHUNK + tid) * 2 + 1];
  __syncthreads();
  for (int st = 32; st > 0; st >>= 1) {
    if (tid < st) { s0[tid] += s0[tid + st]; s1[tid] += s1[tid + st]; }
    __syncthreads();
  }
  if (tid == 0) {
    double mu  = s0[0] / (double)count;
    double var = s1[0] / (double)count - mu * mu;
    if (var < 0.0) var = 0.0;
    stats[2 * batch + 0] = (float)mu;
    stats[2 * batch + 1] = rsqrtf((float)var + 1e-5f);
  }
}

__global__ __launch_bounds__(256) void norm_softmax_rows(
    float* __restrict__ x, int rowLen, int rowsPerStat, const float* __restrict__ stats)
{
  __shared__ float buf[3072];
  __shared__ float red[256];
  const long long row = blockIdx.x;
  x += row * (long long)rowLen;
  const int st = (int)(row / rowsPerStat);
  const float mu = stats[2 * st], rs = stats[2 * st + 1];
  const int tid = threadIdx.x;

  float lmax = -3.4e38f;
  for (int i = tid; i < rowLen; i += 256) {
    float v = (x[i] - mu) * rs;
    buf[i] = v; lmax = fmaxf(lmax, v);
  }
  red[tid] = lmax; __syncthreads();
  for (int s = 128; s > 0; s >>= 1) {
    if (tid < s) red[tid] = fmaxf(red[tid], red[tid + s]);
    __syncthreads();
  }
  const float m = red[0]; __syncthreads();

  float lsum = 0.f;
  for (int i = tid; i < rowLen; i += 256) {
    float e = expf(buf[i] - m);
    buf[i] = e; lsum += e;
  }
  red[tid] = lsum; __syncthreads();
  for (int s = 128; s > 0; s >>= 1) {
    if (tid < s) red[tid] += red[tid + s];
    __syncthreads();
  }
  const float inv = 1.f / red[0];
  for (int i = tid; i < rowLen; i += 256) x[i] = buf[i] * inv;
}

// ---------------------------------------------------------------------------
// Pass 1: score tiles via WMMA, reduced to (sum, sumsq) partials only.
// grid (NBX=24 kv-tiles, NBY=16 q-tiles, 16 bh);  no attn materialization.
// ---------------------------------------------------------------------------
__global__ __launch_bounds__(256) void scores_stat(
    const float* __restrict__ Qg, const float* __restrict__ Kg, float2* __restrict__ partials)
{
  __shared__ __align__(16) unsigned short Qsh[64][72];
  __shared__ __align__(16) unsigned short Ksh[128][72];
  __shared__ float r0[256], r1[256];

  const int tid = threadIdx.x;
  const int z = blockIdx.z, b = z >> 3, h = z & 7;
  const float* Qb = Qg + (long long)b * NEc  + h * 64 + (long long)blockIdx.y * 64  * EEc;
  const float* Kb = Kg + (long long)b * T3Ec + h * 64 + (long long)blockIdx.x * 128 * EEc;

  #pragma unroll
  for (int i = 0; i < 4; ++i) {
    int idx = tid + i * 256; int r = idx >> 4, c4 = idx & 15;
    float4 v = *(const float4*)(Qb + (long long)r * EEc + c4 * 4);
    uint2 u; u.x = pack2(v.x, v.y); u.y = pack2(v.z, v.w);
    *(uint2*)&Qsh[r][c4 * 4] = u;
  }
  #pragma unroll
  for (int i = 0; i < 8; ++i) {
    int idx = tid + i * 256; int r = idx >> 4, c4 = idx & 15;
    float4 v = *(const float4*)(Kb + (long long)r * EEc + c4 * 4);
    uint2 u; u.x = pack2(v.x, v.y); u.y = pack2(v.z, v.w);
    *(uint2*)&Ksh[r][c4 * 4] = u;
  }
  __syncthreads();

  const int wid = tid >> 5, lane = tid & 31, lr = lane & 15, half = lane >> 4;
  const int wm = (wid >> 2) * 32, wn = (wid & 3) * 32;
  const v8f vzero = {0.f,0.f,0.f,0.f,0.f,0.f,0.f,0.f};
  v8f acc[2][2];
  acc[0][0] = vzero; acc[0][1] = vzero; acc[1][0] = vzero; acc[1][1] = vzero;

  #pragma unroll
  for (int s2 = 0; s2 < 2; ++s2) {
    const int kc = s2 * 32;
    v16bf a[2], bb[2];
    #pragma unroll
    for (int i = 0; i < 2; ++i) {
      int row = wm + i * 16 + lr;
      FragBits f;
      f.lo = *(const uint32x4*)&Qsh[row][kc + half * 8];
      f.hi = *(const uint32x4*)&Qsh[row][kc + 16 + half * 8];
      a[i] = __builtin_bit_cast(v16bf, f);
    }
    #pragma unroll
    for (int j = 0; j < 2; ++j) {
      int row = wn + j * 16 + lr;
      FragBits f;
      f.lo = *(const uint32x4*)&Ksh[row][kc + half * 8];
      f.hi = *(const uint32x4*)&Ksh[row][kc + 16 + half * 8];
      bb[j] = __builtin_bit_cast(v16bf, f);
    }
    #pragma unroll
    for (int i = 0; i < 2; ++i)
      #pragma unroll
      for (int j = 0; j < 2; ++j)
        acc[i][j] = __builtin_amdgcn_wmma_f32_16x16x32_bf16(
            false, a[i], false, bb[j], (short)0, acc[i][j], false, false);
  }

  float s = 0.f, ss = 0.f;
  #pragma unroll
  for (int i = 0; i < 2; ++i)
    #pragma unroll
    for (int j = 0; j < 2; ++j)
      #pragma unroll
      for (int r = 0; r < 8; ++r) { float v = acc[i][j][r]; s += v; ss += v * v; }
  r0[tid] = s; r1[tid] = ss; __syncthreads();
  for (int st = 128; st > 0; st >>= 1) {
    if (tid < st) { r0[tid] += r0[tid + st]; r1[tid] += r1[tid + st]; }
    __syncthreads();
  }
  if (tid == 0) {
    float2 p; p.x = r0[0]; p.y = r1[0];
    partials[(long long)z * NPART + blockIdx.y * NBX + blockIdx.x] = p;
  }
}

__global__ __launch_bounds__(256) void stat_final2(
    const float2* __restrict__ partials, float* __restrict__ stats)
{
  __shared__ double s0[256], s1[256];
  const int z = blockIdx.x, tid = threadIdx.x;
  double a = 0.0, b = 0.0;
  for (int i = tid; i < NPART; i += 256) {
    float2 p = partials[(long long)z * NPART + i];
    a += (double)p.x; b += (double)p.y;
  }
  s0[tid] = a; s1[tid] = b; __syncthreads();
  for (int st = 128; st > 0; st >>= 1) {
    if (tid < st) { s0[tid] += s0[tid + st]; s1[tid] += s1[tid + st]; }
    __syncthreads();
  }
  if (tid == 0) {
    const double count = (double)NT3c;
    double mu  = s0[0] / count;
    double var = s1[0] / count - mu * mu;
    if (var < 0.0) var = 0.0;
    stats[2 * z + 0] = (float)mu;
    stats[2 * z + 1] = rsqrtf((float)var + 1e-5f);
  }
}

// ---------------------------------------------------------------------------
// Pass 2: flash-style fused  softmax(scores*rstd) @ V  per (b,h), never
// materializing the [N,3N] matrix.  grid (16 q-tiles, 1, 16 bh).
// Block 256 = 8 waves; q-tile 64 rows; kv-tile 128; online softmax in LDS.
// ---------------------------------------------------------------------------
__global__ __launch_bounds__(256) void flash_attn(
    const float* __restrict__ Qg, const float* __restrict__ Kg, const float* __restrict__ Vg,
    float* __restrict__ Og, const float* __restrict__ stats)
{
  __shared__ __align__(16) unsigned short Qsh[64][72];
  __shared__ __align__(16) unsigned short Ksh[128][72];
  __shared__ __align__(16) unsigned short Vsh[64][136];   // V transposed: [dd][t]
  __shared__ __align__(16) unsigned short Psh[64][136];   // exp(P) bf16: [row][t]
  __shared__ float Sbuf[64][132];
  __shared__ float rowM[64], rowL[64], rowScale[64];
  __shared__ float qmax[64][4], qsum[64][4];

  const int tid = threadIdx.x;
  const int z = blockIdx.z, b = z >> 3, h = z & 7;
  const float rs = stats[2 * z + 1];

  const float* Qb = Qg + (long long)b * NEc  + h * 64 + (long long)blockIdx.x * 64 * EEc;
  const float* Kb = Kg + (long long)b * T3Ec + h * 64;
  const float* Vb = Vg + (long long)b * T3Ec + h * 64;
  float*       Ob = Og + (long long)b * NEc  + h * 64 + (long long)blockIdx.x * 64 * EEc;

  const int wid = tid >> 5, lane = tid & 31, lr = lane & 15, half = lane >> 4;
  const int wm = (wid >> 2) * 32;          // wave rows (scores & PV)
  const int sn = (wid & 3) * 32;           // scores cols
  const int pn = (wid & 3) * 16;           // PV cols (head dim)
  const int srow = tid >> 2, sq = tid & 3; // softmax: 4 threads per row

  // stage Q (64x64), init row state
  #pragma unroll
  for (int i = 0; i < 4; ++i) {
    int idx = tid + i * 256; int r = idx >> 4, c4 = idx & 15;
    float4 v = *(const float4*)(Qb + (long long)r * EEc + c4 * 4);
    uint2 u; u.x = pack2(v.x, v.y); u.y = pack2(v.z, v.w);
    *(uint2*)&Qsh[r][c4 * 4] = u;
  }
  if (tid < 64) { rowM[tid] = -3.4e38f; rowL[tid] = 0.f; }

  const v8f vzero = {0.f,0.f,0.f,0.f,0.f,0.f,0.f,0.f};
  v8f oacc[2]; oacc[0] = vzero; oacc[1] = vzero;
  __syncthreads();

  for (int t0 = 0; t0 < T3c; t0 += 128) {
    // ---- stage K tile (128x64) and V^T tile (64x128) ----
    #pragma unroll
    for (int i = 0; i < 8; ++i) {
      int idx = tid + i * 256; int r = idx >> 4, c4 = idx & 15;
      float4 v = *(const float4*)(Kb + (long long)(t0 + r) * EEc + c4 * 4);
      uint2 u; u.x = pack2(v.x, v.y); u.y = pack2(v.z, v.w);
      *(uint2*)&Ksh[r][c4 * 4] = u;
    }
    #pragma unroll
    for (int i = 0; i < 32; ++i) {
      int idx = tid + i * 256; int dd = idx & 63, t = idx >> 6;
      Vsh[dd][t] = f2bf(Vb[(long long)(t0 + t) * EEc + dd]);
    }
    __syncthreads();

    // ---- scores S = Q x K^T (wave: 32x32) ----
    v8f sacc[2][2];
    sacc[0][0] = vzero; sacc[0][1] = vzero; sacc[1][0] = vzero; sacc[1][1] = vzero;
    #pragma unroll
    for (int s2 = 0; s2 < 2; ++s2) {
      const int kc = s2 * 32;
      v16bf a[2], bb[2];
      #pragma unroll
      for (int i = 0; i < 2; ++i) {
        int row = wm + i * 16 + lr;
        FragBits f;
        f.lo = *(const uint32x4*)&Qsh[row][kc + half * 8];
        f.hi = *(const uint32x4*)&Qsh[row][kc + 16 + half * 8];
        a[i] = __builtin_bit_cast(v16bf, f);
      }
      #pragma unroll
      for (int j = 0; j < 2; ++j) {
        int row = sn + j * 16 + lr;
        FragBits f;
        f.lo = *(const uint32x4*)&Ksh[row][kc + half * 8];
        f.hi = *(const uint32x4*)&Ksh[row][kc + 16 + half * 8];
        bb[j] = __builtin_bit_cast(v16bf, f);
      }
      #pragma unroll
      for (int i = 0; i < 2; ++i)
        #pragma unroll
        for (int j = 0; j < 2; ++j)
          sacc[i][j] = __builtin_amdgcn_wmma_f32_16x16x32_bf16(
              false, a[i], false, bb[j], (short)0, sacc[i][j], false, false);
    }
    #pragma unroll
    for (int i = 0; i < 2; ++i)
      #pragma unroll
      for (int j = 0; j < 2; ++j)
        #pragma unroll
        for (int r = 0; r < 8; ++r)
          Sbuf[wm + i * 16 + half * 8 + r][sn + j * 16 + lr] = sacc[i][j][r];
    __syncthreads();

    // ---- online softmax (4 threads per row; 32 cols each) ----
    {
      const int c0 = sq * 32;
      float tmax = -3.4e38f;
      #pragma unroll
      for (int c = 0; c < 32; ++c) tmax = fmaxf(tmax, Sbuf[srow][c0 + c]);
      qmax[srow][sq] = tmax;
      __syncthreads();
      if (sq == 0) {
        float m4 = fmaxf(fmaxf(qmax[srow][0], qmax[srow][1]),
                         fmaxf(qmax[srow][2], qmax[srow][3])) * rs;
        float mold = rowM[srow];
        float mnew = fmaxf(mold, m4);
        rowScale[srow] = expf(mold - mnew);
        rowM[srow] = mnew;
      }
      __syncthreads();
      const float mnew = rowM[srow];
      float tsum = 0.f;
      #pragma unroll
      for (int c = 0; c < 32; ++c) {
        float p = expf(Sbuf[srow][c0 + c] * rs - mnew);
        Psh[srow][c0 + c] = f2bf(p);
        tsum += p;
      }
      qsum[srow][sq] = tsum;
      __syncthreads();
      if (sq == 0)
        rowL[srow] = rowL[srow] * rowScale[srow]
                   + qsum[srow][0] + qsum[srow][1] + qsum[srow][2] + qsum[srow][3];
    }

    // ---- rescale O accumulators, then O += P x V ----
    #pragma unroll
    for (int i = 0; i < 2; ++i)
      #pragma unroll
      for (int r = 0; r < 8; ++r)
        oacc[i][r] *= rowScale[wm + i * 16 + half * 8 + r];

    #pragma unroll
    for (int kc4 = 0; kc4 < 4; ++kc4) {
      const int kc = kc4 * 32;
      v16bf a[2], bb;
      #pragma unroll
      for (int i = 0; i < 2; ++i) {
        int row = wm + i * 16 + lr;
        FragBits f;
        f.lo = *(const uint32x4*)&Psh[row][kc + half * 8];
        f.hi = *(const uint32x4*)&Psh[row][kc + 16 + half * 8];
        a[i] = __builtin_bit_cast(v16bf, f);
      }
      {
        int row = pn + lr;
        FragBits f;
        f.lo = *(const uint32x4*)&Vsh[row][kc + half * 8];
        f.hi = *(const uint32x4*)&Vsh[row][kc + 16 + half * 8];
        bb = __builtin_bit_cast(v16bf, f);
      }
      oacc[0] = __builtin_amdgcn_wmma_f32_16x16x32_bf16(false, a[0], false, bb, (short)0, oacc[0], false, false);
      oacc[1] = __builtin_amdgcn_wmma_f32_16x16x32_bf16(false, a[1], false, bb, (short)0, oacc[1], false, false);
    }
    __syncthreads();
  }

  // ---- finalize: O /= rowL, store [64 x 64] tile ----
  #pragma unroll
  for (int i = 0; i < 2; ++i)
    #pragma unroll
    for (int r = 0; r < 8; ++r) {
      int row = wm + i * 16 + half * 8 + r;
      Ob[(long long)row * EEc + pn + lr] = oacc[i][r] / rowL[row];
    }
}

// ---------------------------------------------------------------------------
// Host orchestration
// ---------------------------------------------------------------------------
static inline void gemm(hipStream_t st,
    const float* A, long long rsA, long long csA, long long sA1, long long sA2,
    const float* Bm, long long rsB, long long csB, long long sB1, long long sB2,
    float* C, long long ldC, long long sC1, long long sC2,
    int M, int Nn, int K, int nz, int n2, bool va, bool vb)
{
  dim3 g((unsigned)((Nn + 127) / 128), (unsigned)((M + 63) / 64), (unsigned)nz);
  dim3 blk(256);
  if (va && vb)
    gemm_bf16_nt<true, true><<<g, blk, 0, st>>>(A, rsA, csA, sA1, sA2, Bm, rsB, csB, sB1, sB2, C, ldC, sC1, sC2, M, Nn, K, n2);
  else if (vb)
    gemm_bf16_nt<false, true><<<g, blk, 0, st>>>(A, rsA, csA, sA1, sA2, Bm, rsB, csB, sB1, sB2, C, ldC, sC1, sC2, M, Nn, K, n2);
  else
    gemm_bf16_nt<false, false><<<g, blk, 0, st>>>(A, rsA, csA, sA1, sA2, Bm, rsB, csB, sB1, sB2, C, ldC, sC1, sC2, M, Nn, K, n2);
}

extern "C" void kernel_launch(void* const* d_in, const int* in_sizes, int n_in,
                              void* d_out, int out_size, void* d_ws, size_t ws_size,
                              hipStream_t stream)
{
  const int  B = 2, N = 1024, E = 512, H = 8, KVC = 1536;
  const long long NE   = NEc;
  const long long NKVC = (long long)N * KVC;
  const long long KVC2 = (long long)KVC * KVC;

  const float* emb1 = (const float*)d_in[0];
  const float* emb2 = (const float*)d_in[1];
  const float* emb3 = (const float*)d_in[2];
  const float* embC = (const float*)d_in[3];
  const float* Wq1  = (const float*)d_in[4];
  const float* Wq2  = (const float*)d_in[5];
  const float* Wq3  = (const float*)d_in[6];
  const float* Wk   = (const float*)d_in[7];
  const float* Wv   = (const float*)d_in[8];
  const float* WqC  = (const float*)d_in[9];
  const float* WkC  = (const float*)d_in[10];
  const float* WvC  = (const float*)d_in[11];
  const float* Wo1  = (const float*)d_in[12];
  const float* Wo2  = (const float*)d_in[13];
  const float* Wo3  = (const float*)d_in[14];
  float* out = (float*)d_out;

  // workspace arena (float offsets)
  float* ws = (float*)d_ws;
  float*  ctx_c  = ws + 0LL;          // B*KVC*N   = 3,145,728
  float*  Kfull  = ws + 3145728LL;    // B*3N*E    = 3,145,728
  float*  Vfull  = ws + 6291456LL;    // B*3N*E
  float*  Q_C    = ws + 9437184LL;    // B*N*KVC
  float*  K_C    = ws + 12582912LL;
  float*  V_C    = ws + 15728640LL;
  float*  attn_c = ws + 18874368LL;   // B*KVC*KVC = 4,718,592 (ends 23,592,960)
  float*  Qs     = ws + 23592960LL;   // B*N*E
  float*  ctxs   = ws + 24641536LL;   // B*N*E
  double* partD  = (double*)(ws + 25690112LL);  // channel: 2*64*2 doubles
  float2* part2  = (float2*)(ws + 25690624LL);  // spatial: 16*384 float2
  float*  stats  = ws + 25702912LL;             // up to 16 (mu,rstd) pairs

  // ---------------- phase 1: channel attention ----------------
  gemm(stream, embC, KVC, 1, 0, 0,  WqC, KVC, 1, 0, 0,  Q_C, KVC, 0, 0, B*N, KVC, KVC, 1, 1, true, true);
  gemm(stream, embC, KVC, 1, 0, 0,  WkC, KVC, 1, 0, 0,  K_C, KVC, 0, 0, B*N, KVC, KVC, 1, 1, true, true);
  gemm(stream, embC, KVC, 1, 0, 0,  WvC, KVC, 1, 0, 0,  V_C, KVC, 0, 0, B*N, KVC, KVC, 1, 1, true, true);

  // attn_c[b,c,d] = sum_n Q_C[b,n,c] K_C[b,n,d]
  gemm(stream, Q_C, 1, KVC, NKVC, 0,  K_C, 1, KVC, NKVC, 0,
       attn_c, KVC, KVC2, 0, KVC, KVC, N, B, 1, false, false);

  stat_partial<<<dim3(NCHUNK, B), 256, 0, stream>>>(attn_c, KVC2, partD);
  stat_final<<<dim3(B), 64, 0, stream>>>(partD, KVC2, stats);
  norm_softmax_rows<<<dim3((unsigned)(B * KVC)), 256, 0, stream>>>(attn_c, KVC, KVC, stats);

  // ctx_c[b,c,n] = sum_d ch_sim[b,c,d] V_C[b,n,d]
  gemm(stream, attn_c, KVC, 1, KVC2, 0,  V_C, KVC, 1, NKVC, 0,
       ctx_c, N, NKVC, 0, KVC, N, KVC, B, 1, true, true);

  // ---------------- phase 2: shared K/V ----------------
  gemm(stream, ctx_c, 1, N, NE, 0,  Wk, E, 1, 0, 0,  Kfull, E, NE, 0, N, E, E, 3 * B, 1, false, true);
  gemm(stream, ctx_c, 1, N, NE, 0,  Wv, E, 1, 0, 0,  Vfull, E, NE, 0, N, E, E, 3 * B, 1, false, true);

  // ---------------- phase 3: three spatial streams (flash) ----------------
  const float* embs[3] = {emb1, emb2, emb3};
  const float* Wqs[3]  = {Wq1, Wq2, Wq3};
  const float* Wos[3]  = {Wo1, Wo2, Wo3};

  for (int s = 0; s < 3; ++s) {
    gemm(stream, embs[s], E, 1, 0, 0,  Wqs[s], E, 1, 0, 0,  Qs, E, 0, 0, B*N, E, E, 1, 1, true, true);

    // pass 1: instance-norm stats of the (never-materialized) score matrix
    scores_stat<<<dim3(NBX, NBY, B * H), 256, 0, stream>>>(Qs, Kfull, part2);
    stat_final2<<<dim3(B * H), 256, 0, stream>>>(part2, stats);

    // pass 2: fused softmax(scores*rstd) @ V  -> ctxs [B,N,E]
    flash_attn<<<dim3(NBY, 1, B * H), 256, 0, stream>>>(Qs, Kfull, Vfull, ctxs, stats);

    gemm(stream, ctxs, E, 1, 0, 0,  Wos[s], E, 1, 0, 0,
         out + (long long)s * B * NE, E, 0, 0, B*N, E, E, 1, 1, true, true);
  }
  (void)in_sizes; (void)n_in; (void)out_size; (void)ws_size;
}
